// BEVEncoderLayer_67199058313894
// MI455X (gfx1250) — matrix operand order repaired and checked by reading, not compiled, not run on hardware
//
#include <hip/hip_runtime.h>
#include <hip/hip_bf16.h>

typedef __attribute__((ext_vector_type(16))) _Float16 v16h;
typedef __attribute__((ext_vector_type(8)))  _Float16 v8h;
typedef __attribute__((ext_vector_type(8)))  float    v8f;

#define LQn 4096
#define Dn  256
#define Hn  8
#define DHn 32
#define LVn 5376
#define DFFNn 512
#define FA_WAVES 4

#define WMMA_F16(a, b, c) \
    __builtin_amdgcn_wmma_f32_16x16x32_f16(false, (a), false, (b), (short)0, (c), false, false)

// gfx1250 async copy engine: global -> LDS, tracked by ASYNCcnt (in-order completion)
__device__ __forceinline__ void async_b128(void* lds, const _Float16* g) {
    uint32_t l = (uint32_t)(size_t)lds;   // low 32 bits of generic LDS pointer == LDS byte address
    asm volatile("global_load_async_to_lds_b128 %0, %1, off" :: "v"(l), "v"(g) : "memory");
}
#define S_WAIT_ASYNC(N) asm volatile("s_wait_asynccnt " #N ::: "memory")

// ---------------------------------------------------------------------------
// Elementwise helpers
// ---------------------------------------------------------------------------
__global__ void cvt_f16_kernel(const float* __restrict__ s, _Float16* __restrict__ d, int n) {
    int i = blockIdx.x * blockDim.x + threadIdx.x;
    if (i < n) d[i] = (_Float16)s[i];
}

__global__ void add_cvt_f16_kernel(const float* __restrict__ a, const float* __restrict__ b,
                                   _Float16* __restrict__ d, int n) {
    int i = blockIdx.x * blockDim.x + threadIdx.x;
    if (i < n) d[i] = (_Float16)(a[i] + b[i]);
}

__global__ void transpose_f16_kernel(const _Float16* __restrict__ in, _Float16* __restrict__ out,
                                     int R, int C) {
    int i = blockIdx.x * blockDim.x + threadIdx.x;
    if (i < R * C) {
        int r = i / C, c = i % C;
        out[(size_t)c * R + r] = in[i];
    }
}

// ---------------------------------------------------------------------------
// WMMA GEMM: out[M,N] = A[M,K](f16) @ W[N,K](f16)^T + bias
// One wave per 32x32 output (2x2 register-blocked 16x16 WMMA tiles):
// 4 WMMAs per K-step on 4KB of loads (2x the FLOP/byte of a 1x1 tile),
// 4 independent WMMAs back-to-back for XDL pipelining.
// ---------------------------------------------------------------------------
__device__ __forceinline__ v16h load_amat(const _Float16* base, int kb) {
    v16h r;
    ((v8h*)&r)[0] = *(const v8h*)(base + 2 * kb);
    ((v8h*)&r)[1] = *(const v8h*)(base + 2 * kb + 8);
    return r;
}

__global__ void gemm_wmma_kernel(const _Float16* __restrict__ A, int lda,
                                 const _Float16* __restrict__ W,
                                 const float* __restrict__ bias,
                                 float* __restrict__ outF, _Float16* __restrict__ outH,
                                 int ldo, int M, int N, int K, int relu) {
    int wave = threadIdx.x >> 5;
    int lane = threadIdx.x & 31;
    int ntiles = N >> 5;
    int tile = blockIdx.x * (blockDim.x >> 5) + wave;
    if (tile >= (M >> 5) * ntiles) return;            // wave-uniform: EXEC stays all-1
    int tm = tile / ntiles, tn = tile % ntiles;
    int nr = lane & 15, hi = lane >> 4, kb = hi << 3;

    const _Float16* arow0 = A + (size_t)(tm * 32 + nr) * lda;
    const _Float16* arow1 = arow0 + (size_t)16 * lda;
    const _Float16* wrow0 = W + (size_t)(tn * 32 + nr) * K;
    const _Float16* wrow1 = wrow0 + (size_t)16 * K;
    v8f acc00 = {}, acc01 = {}, acc10 = {}, acc11 = {};
    for (int kk = 0; kk < K; kk += 32) {
        if (kk + 64 <= K) {                            // near-cache prefetch of next K block
            __builtin_prefetch(arow0 + kk + 64, 0, 3);
            __builtin_prefetch(arow1 + kk + 64, 0, 3);
        }
        v16h a0 = load_amat(arow0 + kk, kb);
        v16h a1 = load_amat(arow1 + kk, kb);
        v16h b0 = *(const v16h*)(wrow0 + kk + hi * 16);
        v16h b1 = *(const v16h*)(wrow1 + kk + hi * 16);
        acc00 = WMMA_F16(a0, b0, acc00);
        acc01 = WMMA_F16(a0, b1, acc01);
        acc10 = WMMA_F16(a1, b0, acc10);
        acc11 = WMMA_F16(a1, b1, acc11);
    }
    int c0 = tn * 32 + nr, c1 = c0 + 16;
    float bv0 = bias ? bias[c0] : 0.f;
    float bv1 = bias ? bias[c1] : 0.f;
#pragma unroll
    for (int r = 0; r < 8; r++) {
        int row0 = tm * 32 + (hi << 3) + r, row1 = row0 + 16;
        float v00 = acc00[r] + bv0, v01 = acc01[r] + bv1;
        float v10 = acc10[r] + bv0, v11 = acc11[r] + bv1;
        if (relu) {
            v00 = fmaxf(v00, 0.f); v01 = fmaxf(v01, 0.f);
            v10 = fmaxf(v10, 0.f); v11 = fmaxf(v11, 0.f);
        }
        if (outF) {
            outF[(size_t)row0 * ldo + c0] = v00; outF[(size_t)row0 * ldo + c1] = v01;
            outF[(size_t)row1 * ldo + c0] = v10; outF[(size_t)row1 * ldo + c1] = v11;
        }
        if (outH) {
            outH[(size_t)row0 * ldo + c0] = (_Float16)v00; outH[(size_t)row0 * ldo + c1] = (_Float16)v01;
            outH[(size_t)row1 * ldo + c0] = (_Float16)v10; outH[(size_t)row1 * ldo + c1] = (_Float16)v11;
        }
    }
}

// ---------------------------------------------------------------------------
// Flash attention: block = 16-query tile x 4 heads (4 waves), online softmax.
// K/V tiles double-buffered in LDS via async copy engine (8 async b128 per
// wave per 32-key step, s_wait_asynccnt 8 overlaps copy(i+1) with compute(i)).
// All LDS buffers are wave-private; same-wave LDS ops are in-order (DScnt),
// so no block barriers are needed.
// ---------------------------------------------------------------------------
__device__ __forceinline__ void stage_tile(_Float16* lds, const _Float16* g, size_t gstride, int lane) {
    int j = lane >> 2, co = (lane & 3) << 3;          // 32 rows x 64B, 4 b128 per lane
#pragma unroll
    for (int c = 0; c < 4; c++) {
        int row = (c << 3) + j;
        async_b128(lds + row * 32 + co, g + (size_t)row * gstride + co);
    }
}

__global__ void flash_attn_kernel(const _Float16* __restrict__ Qp, int qs,
                                  const _Float16* __restrict__ Kp, int ks,
                                  const _Float16* __restrict__ Vt,
                                  _Float16* __restrict__ Out, int Lk) {
    __shared__ alignas(32) _Float16 lds_k[FA_WAVES][2][32][32];   // [wave][buf][key][dh]
    __shared__ alignas(32) _Float16 lds_v[FA_WAVES][2][32][32];   // [wave][buf][dh][key]
    __shared__ alignas(32) _Float16 lds_p[FA_WAVES][16][32];      // P bounce C->A layout
    int wave = threadIdx.x >> 5, lane = threadIdx.x & 31;
    int h = blockIdx.y * FA_WAVES + wave;
    int qt = blockIdx.x;
    int nr = lane & 15, hi = lane >> 4, kb = hi << 3;
    const float SCALE = 0.17677669529663687f;   // 1/sqrt(32)

    const _Float16* qrow = Qp + (size_t)(qt * 16 + nr) * qs + h * DHn;
    v16h qa = load_amat(qrow, kb);

    v8f o0 = {}, o1 = {};
    float mrow[8], lrow[8];
#pragma unroll
    for (int r = 0; r < 8; r++) { mrow[r] = -1e30f; lrow[r] = 0.f; }

    const int nIter = Lk / 32;
    // prologue: stage tile 0 into buffer 0
    stage_tile(&lds_k[wave][0][0][0], Kp + 0 * ks + h * DHn, ks, lane);
    stage_tile(&lds_v[wave][0][0][0], Vt + (size_t)(h * DHn) * Lk + 0, Lk, lane);

    for (int i = 0; i < nIter; i++) {
        int buf = i & 1;
        if (i + 1 < nIter) {    // issue next-tile copies, then drain current group
            int k1 = (i + 1) * 32;
            stage_tile(&lds_k[wave][buf ^ 1][0][0], Kp + (size_t)k1 * ks + h * DHn, ks, lane);
            stage_tile(&lds_v[wave][buf ^ 1][0][0], Vt + (size_t)(h * DHn) * Lk + k1, Lk, lane);
            S_WAIT_ASYNC(8);    // in-order completion: current 8 copies done
        } else {
            S_WAIT_ASYNC(0);
        }

        const _Float16* kt = &lds_k[wave][buf][0][0];
        v16h b0 = *(const v16h*)(kt + nr * 32 + hi * 16);
        v16h b1 = *(const v16h*)(kt + (16 + nr) * 32 + hi * 16);
        v8f z = {};
        v8f s0 = WMMA_F16(qa, b0, z);
        v8f s1 = WMMA_F16(qa, b1, z);

        float tm[8], ts[8], al[8];
        v8f p0, p1;
#pragma unroll
        for (int r = 0; r < 8; r++) {
            s0[r] *= SCALE; s1[r] *= SCALE;
            tm[r] = fmaxf(s0[r], s1[r]);
        }
#pragma unroll
        for (int o = 8; o >= 1; o >>= 1)
#pragma unroll
            for (int r = 0; r < 8; r++) tm[r] = fmaxf(tm[r], __shfl_xor(tm[r], o, 16));
#pragma unroll
        for (int r = 0; r < 8; r++) {
            float mn = fmaxf(mrow[r], tm[r]);
            al[r] = __expf(mrow[r] - mn);
            mrow[r] = mn;
            p0[r] = __expf(s0[r] - mn);
            p1[r] = __expf(s1[r] - mn);
            ts[r] = p0[r] + p1[r];
        }
#pragma unroll
        for (int o = 8; o >= 1; o >>= 1)
#pragma unroll
            for (int r = 0; r < 8; r++) ts[r] += __shfl_xor(ts[r], o, 16);
#pragma unroll
        for (int r = 0; r < 8; r++) {
            lrow[r] = lrow[r] * al[r] + ts[r];
            o0[r] *= al[r]; o1[r] *= al[r];
            lds_p[wave][(hi << 3) + r][nr]      = (_Float16)p0[r];   // C-layout -> memory
            lds_p[wave][(hi << 3) + r][16 + nr] = (_Float16)p1[r];
        }
        __builtin_amdgcn_wave_barrier();        // same-wave LDS is in-order; just pin schedule
        v16h pa;                                // memory -> A-layout
        ((v8h*)&pa)[0] = *(const v8h*)&lds_p[wave][nr][2 * kb];
        ((v8h*)&pa)[1] = *(const v8h*)&lds_p[wave][nr][2 * kb + 8];
        const _Float16* vt = &lds_v[wave][buf][0][0];
        v16h vb0 = *(const v16h*)(vt + nr * 32 + hi * 16);
        v16h vb1 = *(const v16h*)(vt + (16 + nr) * 32 + hi * 16);
        o0 = WMMA_F16(pa, vb0, o0);
        o1 = WMMA_F16(pa, vb1, o1);
        __builtin_amdgcn_wave_barrier();
    }
#pragma unroll
    for (int r = 0; r < 8; r++) {
        float inv = 1.f / lrow[r];
        int row = qt * 16 + (hi << 3) + r;
        Out[(size_t)row * Dn + h * DHn + nr]      = (_Float16)(o0[r] * inv);
        Out[(size_t)row * Dn + h * DHn + 16 + nr] = (_Float16)(o1[r] * inv);
    }
}

// ---------------------------------------------------------------------------
// Residual + LayerNorm (wave per row, 8 rows/block)
// ---------------------------------------------------------------------------
__global__ void resid_ln_kernel(const float* __restrict__ res, const float* __restrict__ del,
                                const float* __restrict__ g, const float* __restrict__ b,
                                float* __restrict__ out) {
    int wave = threadIdx.x >> 5, lane = threadIdx.x & 31;
    int row = blockIdx.x * 8 + wave;
    if (row >= LQn) return;
    const float* xr = res + (size_t)row * Dn;
    const float* dr = del + (size_t)row * Dn;
    float v[8], s = 0.f, ss = 0.f;
#pragma unroll
    for (int j = 0; j < 8; j++) {
        int c = lane + j * 32;
        v[j] = xr[c] + dr[c];
        s += v[j]; ss += v[j] * v[j];
    }
#pragma unroll
    for (int o = 16; o >= 1; o >>= 1) { s += __shfl_xor(s, o, 32); ss += __shfl_xor(ss, o, 32); }
    float mean = s * (1.f / 256.f);
    float var  = ss * (1.f / 256.f) - mean * mean;
    float rstd = rsqrtf(var + 1e-5f);
    float* orow = out + (size_t)row * Dn;
#pragma unroll
    for (int j = 0; j < 8; j++) {
        int c = lane + j * 32;
        orow[c] = (v[j] - mean) * rstd * g[c] + b[c];
    }
}

// ---------------------------------------------------------------------------
// MS-deformable sampling: block = query, wave = head, lane = channel d.
// ---------------------------------------------------------------------------
__global__ void deform_sample_kernel(const float* __restrict__ off, const float* __restrict__ awl,
                                     const float* __restrict__ ref, const float* __restrict__ value,
                                     _Float16* __restrict__ out) {
    int q = blockIdx.x;
    int h = threadIdx.x >> 5, lane = threadIdx.x & 31;

    const float* al = awl + (size_t)q * 96 + h * 12;
    float logits[12], mx = -1e30f, sum = 0.f;
#pragma unroll
    for (int j = 0; j < 12; j++) { logits[j] = al[j]; mx = fmaxf(mx, logits[j]); }
#pragma unroll
    for (int j = 0; j < 12; j++) { logits[j] = __expf(logits[j] - mx); sum += logits[j]; }
    float invs = 1.f / sum;

    const float* offq = off + (size_t)q * 192 + h * 24;
    const int starts[3] = {0, 4096, 5120};
    const int dims[3]   = {64, 32, 16};
    float acc = 0.f;
#pragma unroll
    for (int l = 0; l < 3; l++) {
        float rx = ref[(size_t)q * 6 + l * 2 + 0];
        float ry = ref[(size_t)q * 6 + l * 2 + 1];
        int Wl = dims[l], Hl = dims[l], st = starts[l];
        float fW = (float)Wl, fH = (float)Hl;
#pragma unroll
        for (int p = 0; p < 4; p++) {
            float ox = offq[l * 8 + p * 2 + 0];
            float oy = offq[l * 8 + p * 2 + 1];
            float x = (rx + ox / fW) * fW - 0.5f;
            float y = (ry + oy / fH) * fH - 0.5f;
            float x0f = floorf(x), y0f = floorf(y);
            float wx = x - x0f, wy = y - y0f;
            int x0 = (int)x0f, y0 = (int)y0f;
            float a = logits[l * 4 + p] * invs;
            float samp = 0.f;
#pragma unroll
            for (int cy = 0; cy < 2; cy++)
#pragma unroll
                for (int cx = 0; cx < 2; cx++) {
                    int xi = x0 + cx, yi = y0 + cy;
                    float w = (cx ? wx : 1.f - wx) * (cy ? wy : 1.f - wy);
                    if (xi >= 0 && xi < Wl && yi >= 0 && yi < Hl)
                        samp += w * value[(size_t)(st + yi * Wl + xi) * Dn + h * DHn + lane];
                }
            acc += a * samp;
        }
    }
    out[(size_t)q * Dn + h * DHn + lane] = (_Float16)acc;
}

// ---------------------------------------------------------------------------
// Host orchestration
// ---------------------------------------------------------------------------
extern "C" void kernel_launch(void* const* d_in, const int* in_sizes, int n_in,
                              void* d_out, int out_size, void* d_ws, size_t ws_size,
                              hipStream_t stream) {
    (void)in_sizes; (void)n_in; (void)out_size; (void)ws_size;
    const float* bev   = (const float*)d_in[0];
    const float* img   = (const float*)d_in[1];
    const float* refp  = (const float*)d_in[2];
    const float* pos   = (const float*)d_in[4];
    const float* miw   = (const float*)d_in[5];
    const float* mib   = (const float*)d_in[6];
    const float* mow   = (const float*)d_in[7];
    const float* mob   = (const float*)d_in[8];
    const float* n1g   = (const float*)d_in[9];
    const float* n1b   = (const float*)d_in[10];
    const float* offw  = (const float*)d_in[11];
    const float* offb  = (const float*)d_in[12];
    const float* attw  = (const float*)d_in[13];
    const float* attb  = (const float*)d_in[14];
    const float* valw  = (const float*)d_in[15];
    const float* valb  = (const float*)d_in[16];
    const float* caw   = (const float*)d_in[17];
    const float* cab   = (const float*)d_in[18];
    const float* n2g   = (const float*)d_in[19];
    const float* n2b   = (const float*)d_in[20];
    const float* f1w   = (const float*)d_in[21];
    const float* f1b   = (const float*)d_in[22];
    const float* f2w   = (const float*)d_in[23];
    const float* f2b   = (const float*)d_in[24];
    const float* fg    = (const float*)d_in[25];
    const float* fbb   = (const float*)d_in[26];

    char* ws = (char*)d_ws;
    size_t cur = 0;
    auto alloc = [&](size_t bytes) -> void* {
        void* p = ws + cur;
        cur += (bytes + 255) & ~(size_t)255;
        return p;
    };
    _Float16* w_mi  = (_Float16*)alloc((size_t)768 * 256 * 2);
    _Float16* w_mo  = (_Float16*)alloc((size_t)256 * 256 * 2);
    _Float16* w_off = (_Float16*)alloc((size_t)192 * 256 * 2);
    _Float16* w_att = (_Float16*)alloc((size_t)96  * 256 * 2);
    _Float16* w_val = (_Float16*)alloc((size_t)256 * 256 * 2);
    _Float16* w_ca  = (_Float16*)alloc((size_t)256 * 256 * 2);
    _Float16* w_f1  = (_Float16*)alloc((size_t)512 * 256 * 2);
    _Float16* w_f2  = (_Float16*)alloc((size_t)256 * 512 * 2);
    _Float16* q_h    = (_Float16*)alloc((size_t)LQn * Dn * 2);
    _Float16* bev_h  = (_Float16*)alloc((size_t)LQn * Dn * 2);
    _Float16* img_h  = (_Float16*)alloc((size_t)LVn * Dn * 2);
    _Float16* qk_h   = (_Float16*)alloc((size_t)LQn * 512 * 2);
    _Float16* vp_h   = (_Float16*)alloc((size_t)LQn * Dn * 2);
    _Float16* vt_h   = (_Float16*)alloc((size_t)Dn * LQn * 2);
    _Float16* attn_h = (_Float16*)alloc((size_t)LQn * Dn * 2);
    float*    proj_f = (float*)alloc((size_t)LQn * Dn * 4);
    float*    x1_f   = (float*)alloc((size_t)LQn * Dn * 4);
    _Float16* q2_h   = (_Float16*)alloc((size_t)LQn * Dn * 2);
    float*    val_f  = (float*)alloc((size_t)LVn * Dn * 4);
    float*    off_f  = (float*)alloc((size_t)LQn * 192 * 4);
    float*    awl_f  = (float*)alloc((size_t)LQn * 96 * 4);
    _Float16* def_h  = (_Float16*)alloc((size_t)LQn * Dn * 2);
    float*    x2_f   = (float*)alloc((size_t)LQn * Dn * 4);
    _Float16* x2_h   = (_Float16*)alloc((size_t)LQn * Dn * 2);
    _Float16* h1_h   = (_Float16*)alloc((size_t)LQn * DFFNn * 2);

    auto cvt = [&](const float* s, _Float16* d, int n) {
        cvt_f16_kernel<<<(n + 255) / 256, 256, 0, stream>>>(s, d, n);
    };
    auto gemm = [&](const _Float16* A, int lda, const _Float16* W, const float* bias,
                    float* oF, _Float16* oH, int ldo, int M, int N, int K, int relu) {
        int tiles = (M / 32) * (N / 32);
        gemm_wmma_kernel<<<(tiles + 7) / 8, 256, 0, stream>>>(A, lda, W, bias, oF, oH, ldo, M, N, K, relu);
    };

    // weight conversion
    cvt(miw, w_mi, 768 * 256);   cvt(mow, w_mo, 256 * 256);
    cvt(offw, w_off, 192 * 256); cvt(attw, w_att, 96 * 256);
    cvt(valw, w_val, 256 * 256); cvt(caw, w_ca, 256 * 256);
    cvt(f1w, w_f1, 512 * 256);   cvt(f2w, w_f2, 256 * 512);
    // activations
    add_cvt_f16_kernel<<<(LQn * Dn + 255) / 256, 256, 0, stream>>>(bev, pos, q_h, LQn * Dn);
    cvt(bev, bev_h, LQn * Dn);
    cvt(img, img_h, LVn * Dn);

    // ---- MHA ----
    gemm(q_h, Dn, w_mi, mib, nullptr, qk_h, 512, LQn, 512, 256, 0);              // Q|K proj
    gemm(bev_h, Dn, w_mi + (size_t)512 * 256, mib + 512, nullptr, vp_h, Dn,      // V proj
         LQn, 256, 256, 0);
    transpose_f16_kernel<<<(LQn * Dn + 255) / 256, 256, 0, stream>>>(vp_h, vt_h, LQn, Dn);
    flash_attn_kernel<<<dim3(LQn / 16, Hn / FA_WAVES), 32 * FA_WAVES, 0, stream>>>(
        qk_h, 512, qk_h + 256, 512, vt_h, attn_h, LQn);
    gemm(attn_h, Dn, w_mo, mob, proj_f, nullptr, Dn, LQn, 256, 256, 0);          // out proj
    resid_ln_kernel<<<LQn / 8, 256, 0, stream>>>(bev, proj_f, n1g, n1b, x1_f);

    // ---- Deformable cross-attention ----
    add_cvt_f16_kernel<<<(LQn * Dn + 255) / 256, 256, 0, stream>>>(x1_f, pos, q2_h, LQn * Dn);
    gemm(img_h, Dn, w_val, valb, val_f, nullptr, Dn, LVn, 256, 256, 0);          // value proj
    gemm(q2_h, Dn, w_off, offb, off_f, nullptr, 192, LQn, 192, 256, 0);          // offsets
    gemm(q2_h, Dn, w_att, attb, awl_f, nullptr, 96, LQn, 96, 256, 0);            // attn logits
    deform_sample_kernel<<<LQn, 256, 0, stream>>>(off_f, awl_f, refp, val_f, def_h);
    gemm(def_h, Dn, w_ca, cab, proj_f, nullptr, Dn, LQn, 256, 256, 0);           // out proj
    resid_ln_kernel<<<LQn / 8, 256, 0, stream>>>(x1_f, proj_f, n2g, n2b, x2_f);

    // ---- FFN ----
    cvt(x2_f, x2_h, LQn * Dn);
    gemm(x2_h, Dn, w_f1, f1b, nullptr, h1_h, DFFNn, LQn, 512, 256, 1);           // fc1 + relu
    gemm(h1_h, DFFNn, w_f2, f2b, proj_f, nullptr, Dn, LQn, 256, 512, 0);         // fc2
    resid_ln_kernel<<<LQn / 8, 256, 0, stream>>>(x2_f, proj_f, fg, fbb, (float*)d_out);
}
